// MySpikeGPT_29394756174185
// MI455X (gfx1250) — matrix operand order repaired and checked by reading, not compiled, not run on hardware
//
#include <hip/hip_runtime.h>
#include <hip/hip_bf16.h>
#include <stdint.h>

// ---------------- model constants ----------------
#define T_STEPS 4
#define BATCH   2
#define SEQ     256
#define DMODEL  768
#define HID     3072
#define VOCAB   32000
#define LAYERS  6
#define BS      (BATCH * SEQ)          // 512 rows per timestep
#define ROWS    (T_STEPS * BS)         // 2048 GEMM rows

typedef __attribute__((ext_vector_type(16))) __bf16 v16bf;
typedef __attribute__((ext_vector_type(8)))  __bf16 v8bf;
typedef __attribute__((ext_vector_type(8)))  float  v8f;

// ---------------- helpers ----------------
__device__ __forceinline__ float block_reduce_sum_256(float x, float* red) {
    int tid = threadIdx.x;
    red[tid] = x;
    __syncthreads();
    #pragma unroll
    for (int off = 128; off > 0; off >>= 1) {
        if (tid < off) red[tid] += red[tid + off];
        __syncthreads();
    }
    float r = red[0];
    __syncthreads();
    return r;
}

// ---------------- fp32 -> bf16 weight conversion ----------------
__global__ __launch_bounds__(256) void cvt_bf16_kernel(const float* __restrict__ src,
                                                       __bf16* __restrict__ dst,
                                                       size_t n) {
    size_t i = (size_t)blockIdx.x * 256 + threadIdx.x;
    if (i < n) dst[i] = (__bf16)src[i];
}

// ---------------- encode: emb[x] + sincos PE, LayerNorm, broadcast over T ----------------
__global__ __launch_bounds__(256) void encode_kernel(const int* __restrict__ x,
                                                     const float* __restrict__ emb,
                                                     const float* __restrict__ g,
                                                     const float* __restrict__ bt,
                                                     float* __restrict__ outb) {
    __shared__ float red[256];
    int row = blockIdx.x;                  // b*SEQ + s
    int s   = row % SEQ;
    int tid = threadIdx.x;
    int tok = x[row];
    float e[3];
    float sum = 0.f;
    #pragma unroll
    for (int k = 0; k < 3; ++k) {
        int d = tid + k * 256;
        float expo = (float)(d & ~1) * (1.0f / (float)DMODEL);
        float ang  = (float)s * __powf(10000.0f, -expo);
        float pe   = (d & 1) ? __cosf(ang) : __sinf(ang);
        e[k] = emb[(size_t)tok * DMODEL + d] + pe;
        sum += e[k];
    }
    float m = block_reduce_sum_256(sum, red) * (1.0f / (float)DMODEL);
    float sq = 0.f;
    #pragma unroll
    for (int k = 0; k < 3; ++k) { float dlt = e[k] - m; sq += dlt * dlt; }
    float var = block_reduce_sum_256(sq, red) * (1.0f / (float)DMODEL);
    float rs  = rsqrtf(var + 1e-5f);
    #pragma unroll
    for (int k = 0; k < 3; ++k) {
        int d = tid + k * 256;
        float val = (e[k] - m) * rs * g[d] + bt[d];
        #pragma unroll
        for (int t = 0; t < T_STEPS; ++t)
            outb[((size_t)t * BS + row) * DMODEL + d] = val;
    }
}

// ---------------- IF neuron over T (elementwise), emits bf16 spikes ----------------
__global__ __launch_bounds__(256) void if_spike_kernel(const float* __restrict__ xin,
                                                       __bf16* __restrict__ spk,
                                                       size_t n /* per-timestep elems */) {
    size_t i = (size_t)blockIdx.x * 256 + threadIdx.x;
    if (i >= n) return;
    float v = 0.f;
    #pragma unroll
    for (int t = 0; t < T_STEPS; ++t) {
        v += xin[(size_t)t * n + i];
        float sp = (v >= 1.0f) ? 1.0f : 0.0f;
        v *= (1.0f - sp);
        spk[(size_t)t * n + i] = (__bf16)sp;
    }
}

// ---------------- bf16 WMMA GEMM:  C[M,N] = A[M,K] * B[K,N]  (fp32 accum) ----------------
// Compile-time N,K so every address folds to base + imm offset.
// Block tile 128x128, 8 wave32 waves (4 along M x 2 along N), each wave 32x64.
// BK=64: 16 v_wmma per iteration. Double-buffered LDS, async-DMA A tiles,
// uint4-register-prefetched B tiles packed via v_perm, one barrier per K-step,
// peeled steady-state pipeline (no branches in the hot loop).
#define BM 128
#define BN 128
#define BK 64

template <int N, int K>
__global__ __launch_bounds__(256) void gemm_bf16_kernel(const __bf16* __restrict__ A,
                                                        const __bf16* __restrict__ Bw,
                                                        float* __restrict__ C) {
    constexpr int NIT = K / BK;
    __shared__ __bf16   Asm[2][BM][BK];          // [stage][m][k]
    __shared__ uint32_t Bsm[2][BN][BK / 2];      // [stage][n][k/2]  (2 bf16 packed per u32)

    const int tid  = threadIdx.x;
    const int lane = tid & 31;
    const int wid  = tid >> 5;
    const int wm   = wid & 3;                 // wave row  (4 x 32 rows)
    const int wn   = wid >> 2;                // wave col  (2 x 64 cols)
    const int ln16 = lane & 15;
    const int hi   = lane >> 4;               // half-wave select

    const int bm = blockIdx.y * BM;
    const int bn = blockIdx.x * BN;

    // A staging: each thread async-copies 64 bytes (32 bf16) of one row
    const int ar = tid >> 1;
    const int ac = (tid & 1) * 32;
    // B staging: each thread handles 2 k-rows x 16 n
    const int bk2 = (tid >> 3) * 2;           // 0,2,..,62
    const int bnn = (tid & 7) * 16;           // 0..112

    const __bf16* agp = A + (size_t)(bm + ar) * K + ac;
    const uint4*  bgp = (const uint4*)(Bw + (size_t)bk2 * N + bn + bnn);
    constexpr int BROW4  = N / 8;             // uint4 per B k-row
    constexpr int BSTEP4 = BK * N / 8;        // uint4 advance per K-step

    auto issueA = [&](int stage, int k0) {
        uint64_t ga = (uint64_t)(uintptr_t)(agp + k0);
        uint32_t la = (uint32_t)(uintptr_t)(&Asm[stage][ar][ac]);
        asm volatile("global_load_async_to_lds_b128 %0, %1, off"           :: "v"(la), "v"(ga) : "memory");
        asm volatile("global_load_async_to_lds_b128 %0, %1, off offset:16" :: "v"(la), "v"(ga) : "memory");
        asm volatile("global_load_async_to_lds_b128 %0, %1, off offset:32" :: "v"(la), "v"(ga) : "memory");
        asm volatile("global_load_async_to_lds_b128 %0, %1, off offset:48" :: "v"(la), "v"(ga) : "memory");
    };

    // B prefetch registers: two k-rows x 32 bytes, guaranteed b128 global loads
    uint4 q00, q01, q10, q11;
    auto loadB = [&](int it) {
        const uint4* p = bgp + (size_t)it * BSTEP4;
        q00 = p[0];
        q01 = p[1];
        q10 = p[BROW4];
        q11 = p[BROW4 + 1];
    };
    // transpose-pack into LDS: pk(e) = { r1[e] : r0[e] } (hi:lo bf16)
    auto storeB = [&](int stage) {
        uint32_t r0[8] = {q00.x, q00.y, q00.z, q00.w, q01.x, q01.y, q01.z, q01.w};
        uint32_t r1[8] = {q10.x, q10.y, q10.z, q10.w, q11.x, q11.y, q11.z, q11.w};
        #pragma unroll
        for (int e = 0; e < 16; ++e) {
            uint32_t pk = __builtin_amdgcn_perm(r1[e >> 1], r0[e >> 1],
                                                (e & 1) ? 0x07060302u : 0x05040100u);
            Bsm[stage][bnn + e][bk2 >> 1] = pk;
        }
    };

    v8f acc[2][4];
    #pragma unroll
    for (int i = 0; i < 2; ++i)
        #pragma unroll
        for (int j = 0; j < 4; ++j)
            #pragma unroll
            for (int r = 0; r < 8; ++r) acc[i][j][r] = 0.0f;

    auto compute = [&](int cur) {
        #pragma unroll
        for (int c = 0; c < 2; ++c) {
            v16bf afrag[2];
            #pragma unroll
            for (int i = 0; i < 2; ++i) {
                int row = wm * 32 + i * 16 + ln16;
                v8bf* ap = (v8bf*)&afrag[i];
                ap[0] = *(const v8bf*)&Asm[cur][row][c * 32 + hi * 16];
                ap[1] = *(const v8bf*)&Asm[cur][row][c * 32 + hi * 16 + 8];
            }
            v16bf bfrag[4];
            #pragma unroll
            for (int j = 0; j < 4; ++j) {
                int col = wn * 64 + j * 16 + ln16;
                bfrag[j] = *(const v16bf*)&Bsm[cur][col][c * 16 + hi * 8];
            }
            #pragma unroll
            for (int i = 0; i < 2; ++i)
                #pragma unroll
                for (int j = 0; j < 4; ++j)
                    acc[i][j] = __builtin_amdgcn_wmma_f32_16x16x32_bf16(
                        false, afrag[i], false, bfrag[j],
                        (short)0, acc[i][j], false, false);
        }
    };

    // prologue: stage 0
    issueA(0, 0);
    loadB(0);
    storeB(0);
    asm volatile("s_wait_asynccnt 0x0" ::: "memory");
    __syncthreads();

    // steady state: prefetch next stage, compute current, commit B, sync
    for (int it = 0; it < NIT - 1; ++it) {
        const int cur = it & 1;
        const int nxt = cur ^ 1;
        issueA(nxt, (it + 1) * BK);
        loadB(it + 1);
        compute(cur);
        storeB(nxt);
        asm volatile("s_wait_asynccnt 0x0" ::: "memory");
        __syncthreads();
    }
    // final iteration: compute only
    compute((NIT - 1) & 1);

    // ---- store C: constant strides fold into immediate offsets ----
    float* cp = C + (size_t)(bm + wm * 32 + hi * 8) * N + (bn + wn * 64 + ln16);
    #pragma unroll
    for (int i = 0; i < 2; ++i)
        #pragma unroll
        for (int j = 0; j < 4; ++j)
            #pragma unroll
            for (int r = 0; r < 8; ++r)
                cp[(size_t)(i * 16 + r) * N + j * 16] = acc[i][j][r];
}

// ---------------- LayerNorm per (t,row) then IF over T, emits bf16 spikes ----------------
template <int DPT>
__global__ __launch_bounds__(256) void ln_if_kernel(const float* __restrict__ y,
                                                    const float* __restrict__ g,
                                                    const float* __restrict__ bb,
                                                    __bf16* __restrict__ spk) {
    __shared__ float red[256];
    const int Dd = DPT * 256;
    int row = blockIdx.x;               // 0..BS-1
    int tid = threadIdx.x;
    float v[DPT];
    #pragma unroll
    for (int k = 0; k < DPT; ++k) v[k] = 0.f;

    for (int t = 0; t < T_STEPS; ++t) {
        size_t base = ((size_t)t * BS + row) * Dd;
        float xv[DPT];
        float sum = 0.f;
        #pragma unroll
        for (int k = 0; k < DPT; ++k) { xv[k] = y[base + tid + k * 256]; sum += xv[k]; }
        float m = block_reduce_sum_256(sum, red) / (float)Dd;
        float sq = 0.f;
        #pragma unroll
        for (int k = 0; k < DPT; ++k) { float d = xv[k] - m; sq += d * d; }
        float var = block_reduce_sum_256(sq, red) / (float)Dd;
        float rs  = rsqrtf(var + 1e-5f);
        #pragma unroll
        for (int k = 0; k < DPT; ++k) {
            int d = tid + k * 256;
            float nrm = (xv[k] - m) * rs * g[d] + bb[d];
            v[k] += nrm;
            float sp = (v[k] >= 1.0f) ? 1.0f : 0.0f;
            v[k] *= (1.0f - sp);
            spk[base + d] = (__bf16)sp;
        }
    }
}

// ---------------- dst = base + LayerNorm(y)  (per (t,row), no IF coupling) ----------------
template <int DPT>
__global__ __launch_bounds__(256) void ln_add_kernel(const float* __restrict__ y,
                                                     const float* __restrict__ g,
                                                     const float* __restrict__ bb,
                                                     const float* __restrict__ basev,
                                                     float* __restrict__ dst) {
    __shared__ float red[256];
    const int Dd = DPT * 256;
    int row = blockIdx.x;               // 0..ROWS-1
    int tid = threadIdx.x;
    size_t base = (size_t)row * Dd;
    float xv[DPT];
    float sum = 0.f;
    #pragma unroll
    for (int k = 0; k < DPT; ++k) { xv[k] = y[base + tid + k * 256]; sum += xv[k]; }
    float m = block_reduce_sum_256(sum, red) / (float)Dd;
    float sq = 0.f;
    #pragma unroll
    for (int k = 0; k < DPT; ++k) { float d = xv[k] - m; sq += d * d; }
    float var = block_reduce_sum_256(sq, red) / (float)Dd;
    float rs  = rsqrtf(var + 1e-5f);
    #pragma unroll
    for (int k = 0; k < DPT; ++k) {
        int d = tid + k * 256;
        float nrm = (xv[k] - m) * rs * g[d] + bb[d];
        dst[base + d] = basev[base + d] + nrm;
    }
}

// ---------------- QK = IF( sum_s Q*K )  (spike AND-reduce over sequence) ----------------
__global__ __launch_bounds__(256) void qk_if_kernel(const __bf16* __restrict__ Qs,
                                                    const __bf16* __restrict__ Ks,
                                                    __bf16* __restrict__ qks) {
    int i = blockIdx.x * 256 + threadIdx.x;        // 0 .. BATCH*DMODEL-1
    if (i >= BATCH * DMODEL) return;
    int b = i / DMODEL, d = i % DMODEL;
    float v = 0.f;
    for (int t = 0; t < T_STEPS; ++t) {
        size_t base = ((size_t)(t * BATCH + b) * SEQ) * DMODEL + d;
        float sum = 0.f;
        for (int s = 0; s < SEQ; ++s)
            sum += (float)Qs[base + (size_t)s * DMODEL] * (float)Ks[base + (size_t)s * DMODEL];
        v += sum;
        float sp = (v >= 1.0f) ? 1.0f : 0.0f;
        v *= (1.0f - sp);
        qks[(size_t)(t * BATCH + b) * DMODEL + d] = (__bf16)sp;
    }
}

// ---------------- QKV = V_spike * QK_spike (broadcast over S) ----------------
__global__ __launch_bounds__(256) void qkv_kernel(const __bf16* __restrict__ Vs,
                                                  const __bf16* __restrict__ qks,
                                                  __bf16* __restrict__ outq) {
    size_t i = (size_t)blockIdx.x * 256 + threadIdx.x;
    if (i >= (size_t)ROWS * DMODEL) return;
    int d  = (int)(i % DMODEL);
    int r  = (int)(i / DMODEL);                    // t*BS + b*SEQ + s
    int t  = r / BS;
    int b  = (r % BS) / SEQ;
    outq[i] = (__bf16)((float)Vs[i] * (float)qks[(size_t)(t * BATCH + b) * DMODEL + d]);
}

// ---------------- LayerNorm over V=32000 per (t,row), in place ----------------
__global__ __launch_bounds__(256) void ln_bigV_kernel(float* __restrict__ y,
                                                      const float* __restrict__ g,
                                                      const float* __restrict__ bb) {
    __shared__ float red[256];
    int row = blockIdx.x;               // 0..ROWS-1
    int tid = threadIdx.x;
    size_t base = (size_t)row * VOCAB;
    float sum = 0.f, sq = 0.f;
    for (int j = tid; j < VOCAB; j += 256) { float x = y[base + j]; sum += x; sq += x * x; }
    float m   = block_reduce_sum_256(sum, red) / (float)VOCAB;
    float msq = block_reduce_sum_256(sq,  red) / (float)VOCAB;
    float rs  = rsqrtf(msq - m * m + 1e-5f);
    for (int j = tid; j < VOCAB; j += 256)
        y[base + j] = (y[base + j] - m) * rs * g[j] + bb[j];
}

// ---------------- IF over T on logits, mean over T -> d_out ----------------
__global__ __launch_bounds__(256) void if_mean_kernel(const float* __restrict__ lg,
                                                      float* __restrict__ outb,
                                                      size_t n /* BATCH*SEQ*VOCAB */) {
    size_t i = (size_t)blockIdx.x * 256 + threadIdx.x;
    if (i >= n) return;
    float v = 0.f, acc = 0.f;
    #pragma unroll
    for (int t = 0; t < T_STEPS; ++t) {
        v += lg[(size_t)t * n + i];
        float sp = (v >= 1.0f) ? 1.0f : 0.0f;
        acc += sp;
        v *= (1.0f - sp);
    }
    outb[i] = acc * 0.25f;
}

// ---------------- row-normalize d_out by its sum over V ----------------
__global__ __launch_bounds__(256) void row_norm_kernel(float* __restrict__ outb) {
    __shared__ float red[256];
    int row = blockIdx.x;               // 0..BS-1
    int tid = threadIdx.x;
    size_t base = (size_t)row * VOCAB;
    float sum = 0.f;
    for (int j = tid; j < VOCAB; j += 256) sum += outb[base + j];
    float tot = block_reduce_sum_256(sum, red);
    float inv = 1.0f / tot;
    for (int j = tid; j < VOCAB; j += 256) outb[base + j] *= inv;
}

// =====================================================================
extern "C" void kernel_launch(void* const* d_in, const int* in_sizes, int n_in,
                              void* d_out, int out_size, void* d_ws, size_t ws_size,
                              hipStream_t stream) {
    (void)in_sizes; (void)n_in; (void)out_size; (void)ws_size;

    const int*   x      = (const int*)  d_in[0];
    const float* emb    = (const float*)d_in[1];
    const float* enc_g  = (const float*)d_in[2];
    const float* enc_b  = (const float*)d_in[3];
    const float* wq     = (const float*)d_in[4];
    const float* wk     = (const float*)d_in[5];
    const float* wv     = (const float*)d_in[6];
    const float* wo     = (const float*)d_in[7];
    const float* lnq_g  = (const float*)d_in[8];
    const float* lnq_b  = (const float*)d_in[9];
    const float* lnk_g  = (const float*)d_in[10];
    const float* lnk_b  = (const float*)d_in[11];
    const float* lnv_g  = (const float*)d_in[12];
    const float* lnv_b  = (const float*)d_in[13];
    const float* lno_g  = (const float*)d_in[14];
    const float* lno_b  = (const float*)d_in[15];
    const float* w1     = (const float*)d_in[16];
    const float* w2     = (const float*)d_in[17];
    const float* f1_g   = (const float*)d_in[18];
    const float* f1_b   = (const float*)d_in[19];
    const float* f2_g   = (const float*)d_in[20];
    const float* f2_b   = (const float*)d_in[21];
    const float* out_w  = (const float*)d_in[22];
    const float* oln_g  = (const float*)d_in[23];
    const float* oln_b  = (const float*)d_in[24];
    float* out = (float*)d_out;

    // ------- workspace carve-out -------
    char* ws = (char*)d_ws;
    size_t off = 0;
    auto alloc = [&](size_t bytes) -> char* {
        char* p = ws + off;
        off = (off + bytes + 255) & ~(size_t)255;
        return p;
    };
    const size_t DD = (size_t)DMODEL * DMODEL;
    const size_t DH = (size_t)DMODEL * HID;
    __bf16* wq_b  = (__bf16*)alloc(LAYERS * DD * 2);
    __bf16* wk_b  = (__bf16*)alloc(LAYERS * DD * 2);
    __bf16* wv_b  = (__bf16*)alloc(LAYERS * DD * 2);
    __bf16* wo_b  = (__bf16*)alloc(LAYERS * DD * 2);
    __bf16* w1_b  = (__bf16*)alloc(LAYERS * DH * 2);
    __bf16* w2_b  = (__bf16*)alloc(LAYERS * DH * 2);
    __bf16* ow_b  = (__bf16*)alloc((size_t)DMODEL * VOCAB * 2);
    float*  out_buf = (float*)alloc((size_t)ROWS * DMODEL * 4);
    float*  h_buf   = (float*)alloc((size_t)ROWS * DMODEL * 4);
    float*  y_buf   = (float*)alloc((size_t)ROWS * HID * 4);     // big enough for H and D outputs
    __bf16* tmp_spk = (__bf16*)alloc((size_t)ROWS * DMODEL * 2);
    __bf16* q_spk   = (__bf16*)alloc((size_t)ROWS * DMODEL * 2);
    __bf16* k_spk   = (__bf16*)alloc((size_t)ROWS * DMODEL * 2);
    __bf16* v_spk   = (__bf16*)alloc((size_t)ROWS * DMODEL * 2);
    __bf16* qk_spk  = (__bf16*)alloc((size_t)T_STEPS * BATCH * DMODEL * 2);
    __bf16* qkv_spk = (__bf16*)alloc((size_t)ROWS * DMODEL * 2);
    __bf16* h_spk   = (__bf16*)alloc((size_t)ROWS * DMODEL * 2);
    __bf16* f_spk   = (__bf16*)alloc((size_t)ROWS * HID * 2);
    __bf16* o_spk   = (__bf16*)alloc((size_t)ROWS * DMODEL * 2);
    float*  logits  = (float*)alloc((size_t)ROWS * VOCAB * 4);

    auto cvt = [&](const float* s, __bf16* d, size_t n) {
        cvt_bf16_kernel<<<(unsigned)((n + 255) / 256), 256, 0, stream>>>(s, d, n);
    };

    // 1) weights -> bf16
    cvt(wq, wq_b, LAYERS * DD);
    cvt(wk, wk_b, LAYERS * DD);
    cvt(wv, wv_b, LAYERS * DD);
    cvt(wo, wo_b, LAYERS * DD);
    cvt(w1, w1_b, LAYERS * DH);
    cvt(w2, w2_b, LAYERS * DH);
    cvt(out_w, ow_b, (size_t)DMODEL * VOCAB);

    // 2) encode
    encode_kernel<<<BS, 256, 0, stream>>>(x, emb, enc_g, enc_b, out_buf);

    const size_t nTD = (size_t)BS * DMODEL;                 // per-timestep elems (D)
    const unsigned gTD = (unsigned)((nTD + 255) / 256);
    const unsigned gRD = (unsigned)(((size_t)ROWS * DMODEL + 255) / 256);

    const dim3 gridDD(DMODEL / BN, ROWS / BM);   // N=768
    const dim3 gridDH(HID / BN,    ROWS / BM);   // N=3072
    const dim3 gridDV(VOCAB / BN,  ROWS / BM);   // N=32000

    // 3) layers
    for (int l = 0; l < LAYERS; ++l) {
        // SDSA
        if_spike_kernel<<<gTD, 256, 0, stream>>>(out_buf, tmp_spk, nTD);
        gemm_bf16_kernel<DMODEL, DMODEL><<<gridDD, 256, 0, stream>>>(tmp_spk, wq_b + (size_t)l * DD, y_buf);
        ln_if_kernel<3><<<BS, 256, 0, stream>>>(y_buf, lnq_g + l * DMODEL, lnq_b + l * DMODEL, q_spk);
        gemm_bf16_kernel<DMODEL, DMODEL><<<gridDD, 256, 0, stream>>>(tmp_spk, wk_b + (size_t)l * DD, y_buf);
        ln_if_kernel<3><<<BS, 256, 0, stream>>>(y_buf, lnk_g + l * DMODEL, lnk_b + l * DMODEL, k_spk);
        gemm_bf16_kernel<DMODEL, DMODEL><<<gridDD, 256, 0, stream>>>(tmp_spk, wv_b + (size_t)l * DD, y_buf);
        ln_if_kernel<3><<<BS, 256, 0, stream>>>(y_buf, lnv_g + l * DMODEL, lnv_b + l * DMODEL, v_spk);

        qk_if_kernel<<<(BATCH * DMODEL + 255) / 256, 256, 0, stream>>>(q_spk, k_spk, qk_spk);
        qkv_kernel<<<gRD, 256, 0, stream>>>(v_spk, qk_spk, qkv_spk);

        gemm_bf16_kernel<DMODEL, DMODEL><<<gridDD, 256, 0, stream>>>(qkv_spk, wo_b + (size_t)l * DD, y_buf);
        ln_add_kernel<3><<<ROWS, 256, 0, stream>>>(y_buf, lno_g + l * DMODEL, lno_b + l * DMODEL,
                                                   out_buf, h_buf);
        // FFN
        if_spike_kernel<<<gTD, 256, 0, stream>>>(h_buf, h_spk, nTD);
        gemm_bf16_kernel<HID, DMODEL><<<gridDH, 256, 0, stream>>>(h_spk, w1_b + (size_t)l * DH, y_buf);
        ln_if_kernel<12><<<BS, 256, 0, stream>>>(y_buf, f1_g + l * HID, f1_b + l * HID, f_spk);
        gemm_bf16_kernel<DMODEL, HID><<<gridDD, 256, 0, stream>>>(f_spk, w2_b + (size_t)l * DH, y_buf);
        ln_add_kernel<3><<<ROWS, 256, 0, stream>>>(y_buf, f2_g + l * DMODEL, f2_b + l * DMODEL,
                                                   h_buf, out_buf);
    }

    // 4) output head
    if_spike_kernel<<<gTD, 256, 0, stream>>>(out_buf, o_spk, nTD);
    gemm_bf16_kernel<VOCAB, DMODEL><<<gridDV, 256, 0, stream>>>(o_spk, ow_b, logits);
    ln_bigV_kernel<<<ROWS, 256, 0, stream>>>(logits, oln_g, oln_b);
    const size_t nBSV = (size_t)BATCH * SEQ * VOCAB;
    if_mean_kernel<<<(unsigned)((nBSV + 255) / 256), 256, 0, stream>>>(logits, out, nBSV);
    row_norm_kernel<<<BS, 256, 0, stream>>>(out);
}
